// RNN_43147241455677
// MI455X (gfx1250) — matrix operand order repaired and checked by reading, not compiled, not run on hardware
//
#include <hip/hip_runtime.h>
#include <hip/hip_bf16.h>
#include <math.h>

// ---------------------------------------------------------------------------
// Types for CDNA5 WMMA (wave32, 16x16x32 bf16 -> f32)
// ---------------------------------------------------------------------------
typedef __attribute__((ext_vector_type(16))) __bf16 v16bf;
typedef __attribute__((ext_vector_type(8)))  __bf16 v8bf;
typedef __attribute__((ext_vector_type(8)))  float  v8f;

#define RNN_INPUT   1024
#define RNN_HIDDEN  2048
#define RNN_OUTPUT  1024
#define RNN_COMB    3072      // INPUT + HIDDEN
#define RNN_SEQ     1024
#define RNN_HSTEPS  1023      // h updated for t = 0..1022; out uses c[1023]

#define RNN_BLOCKS  128       // 128 * 16 rows = 2048 hidden rows
#define RNN_THREADS 128       // 4 wave32 waves; each wave covers K-slice of 768

// ---------------------------------------------------------------------------
// bit-level fp32 -> bf16 (round-to-nearest-even)
// ---------------------------------------------------------------------------
__device__ __forceinline__ __bf16 f2bf(float f) {
  union { float f; unsigned u; } v; v.f = f;
  unsigned r = v.u + 0x7FFFu + ((v.u >> 16) & 1u);
  unsigned short h = (unsigned short)(r >> 16);
  __bf16 out; __builtin_memcpy(&out, &h, 2);
  return out;
}

// ---------------------------------------------------------------------------
// generation-counting grid barrier (re-initialized every launch by init kernel,
// so graph replay is deterministic). Release via __threadfence, s_sleep spin.
// ---------------------------------------------------------------------------
__device__ __forceinline__ void grid_barrier(int* cnt, int* gen, int nblk) {
  __syncthreads();
  __threadfence();                       // release this block's h stores
  if (threadIdx.x == 0) {
    int g   = __hip_atomic_load(gen, __ATOMIC_ACQUIRE, __HIP_MEMORY_SCOPE_AGENT);
    int old = __hip_atomic_fetch_add(cnt, 1, __ATOMIC_ACQ_REL, __HIP_MEMORY_SCOPE_AGENT);
    if (old == nblk - 1) {
      __hip_atomic_store(cnt, 0, __ATOMIC_RELAXED, __HIP_MEMORY_SCOPE_AGENT);
      __hip_atomic_fetch_add(gen, 1, __ATOMIC_ACQ_REL, __HIP_MEMORY_SCOPE_AGENT);
    } else {
      while (__hip_atomic_load(gen, __ATOMIC_ACQUIRE, __HIP_MEMORY_SCOPE_AGENT) == g)
        __builtin_amdgcn_s_sleep(2);
    }
  }
  __syncthreads();
}

// ---------------------------------------------------------------------------
// 16-row x 3072-col bf16 GEMV tile via WMMA.
//   A : 16 weight rows in the 16-bit A layout (lane<16 -> K {0..7,16..23},
//       lane>=16 -> K {8..15,24..31} of row lane&15)      -> 2x b128 loads
//   B : activation chunk broadcast into ALL 16 columns: lanes 0..15 load
//       c[k+{0..7,16..23}], lanes 16..31 load c[k+{8..15,24..31}] -> 2x b128 loads
//       (no EXEC masking; all columns of D are identical)
//   D : f32 accumulator; lane0 holds rows 0..7, lane16 rows 8..15 (col 0).
// Each of the 4 waves covers K in [wave*768, wave*768+768); partials reduced
// through LDS. Returns bias-free dot for row (row0 + tid) when tid < 16.
// ---------------------------------------------------------------------------
__device__ __forceinline__ float tile_gemv16(
    const __bf16* __restrict__ W,      // [rows x 3072] row-major bf16
    const __bf16* __restrict__ cvec,   // [3072] bf16 = [x_t | h_t] contiguous
    int row0, int tid, float* red /* [4][16] */)
{
  const int lane  = tid & 31;
  const int wave  = tid >> 5;
  const int arow  = lane & 15;
  const int ahalf = (lane >> 4) ? 8 : 0;    // A-fragment K sub-offset
  const int boff  = (lane >> 4) ? 8 : 0;    // B-fragment K sub-offset

  const __bf16* wbase = W + (size_t)(row0 + arow) * RNN_COMB + ahalf;
  const __bf16* cbase = cvec + boff;
  const int kbase = wave * 768;

  v8f acc = {};
  #pragma unroll 4
  for (int kc = 0; kc < 24; ++kc) {
    const int k = kbase + kc * 32;

    const __bf16* wr = wbase + k;
    v8bf a0 = *reinterpret_cast<const v8bf*>(wr);
    v8bf a1 = *reinterpret_cast<const v8bf*>(wr + 16);

    const __bf16* cp = cbase + k;
    v8bf b0 = *reinterpret_cast<const v8bf*>(cp);
    v8bf b1 = *reinterpret_cast<const v8bf*>(cp + 16);

    v16bf a, b;
    #pragma unroll
    for (int i = 0; i < 8; ++i) {
      a[i] = a0[i]; a[8 + i] = a1[i];
      b[i] = b0[i]; b[8 + i] = b1[i];
    }

    acc = __builtin_amdgcn_wmma_f32_16x16x32_bf16(
        /*neg_a=*/false, a, /*neg_b=*/false, b,
        /*c_mod=*/(short)0, acc, /*reuse_a=*/false, /*reuse_b=*/false);
  }

  // lanes 0..15 hold rows 0..7, lanes 16..31 rows 8..15 (columns identical); lane0 / lane16
  // publish rows 0..7 / 8..15 of this wave's partial to LDS
  if (lane == 0) {
    #pragma unroll
    for (int j = 0; j < 8; ++j) red[wave * 16 + j] = acc[j];
  }
  if (lane == 16) {
    #pragma unroll
    for (int j = 0; j < 8; ++j) red[wave * 16 + 8 + j] = acc[j];
  }
  __syncthreads();
  float s = 0.0f;
  if (tid < 16)
    s = red[0 * 16 + tid] + red[1 * 16 + tid] + red[2 * 16 + tid] + red[3 * 16 + tid];
  __syncthreads();   // protect LDS reuse on next call
  return s;
}

// ---------------------------------------------------------------------------
// Kernels
// ---------------------------------------------------------------------------
__global__ void cvt_f32_bf16(const float* __restrict__ src, __bf16* __restrict__ dst, int n) {
  int i = blockIdx.x * blockDim.x + threadIdx.x;
  if (i < n) dst[i] = f2bf(src[i]);
}

// Fill cbuf[t][0:1024] = bf16(x[t]), zero cbuf[0][1024:3072] (h0), zero
// logits accumulator and barrier state. Runs every launch (graph-safe).
__global__ void init_cbuf(const float* __restrict__ x, __bf16* __restrict__ cbuf,
                          float* outacc, int* bar) {
  int idx = blockIdx.x * blockDim.x + threadIdx.x;        // over 1024*3072
  if (idx < RNN_SEQ * RNN_COMB) {
    int t = idx / RNN_COMB;
    int i = idx - t * RNN_COMB;
    if (i < RNN_INPUT) {
      cbuf[idx] = f2bf(x[(size_t)t * RNN_INPUT + i]);
    } else if (t == 0) {
      cbuf[idx] = f2bf(0.0f);                             // h_0 = 0
    }
  }
  if (idx < RNN_OUTPUT) outacc[idx] = 0.0f;
  if (idx < 2)          bar[idx]    = 0;
}

__global__ void __launch_bounds__(RNN_THREADS, 1)
rnn_persistent(const __bf16* __restrict__ Wh,   // [2048 x 3072] bf16
               const __bf16* __restrict__ Wo,   // [1024 x 3072] bf16
               __bf16* cbuf,                    // [1024 x 3072] bf16: [x_t | h_t]
               const float* __restrict__ bh,    // [2048]
               const float* __restrict__ bo,    // [1024]
               float* outacc,                   // [1024] pre-softmax logits
               int* bar_cnt, int* bar_gen)
{
  __shared__ float red[4 * 16];
  const int tid  = threadIdx.x;
  const int blk  = blockIdx.x;
  const int row0 = blk * 16;

  for (int t = 0; t < RNN_HSTEPS; ++t) {
    const __bf16* cvec = cbuf + (size_t)t * RNN_COMB;
    __bf16*       hnxt = cbuf + (size_t)(t + 1) * RNN_COMB + RNN_INPUT;

    float s = tile_gemv16(Wh, cvec, row0, tid, red);
    if (tid < 16) hnxt[row0 + tid] = f2bf(s + bh[row0 + tid]);

    grid_barrier(bar_cnt, bar_gen, RNN_BLOCKS);
  }

  // out = c[1023] @ Wo.T + bo   (blocks 0..63 -> 1024 rows)
  if (blk < 64) {
    const __bf16* cvec = cbuf + (size_t)(RNN_SEQ - 1) * RNN_COMB;
    float s = tile_gemv16(Wo, cvec, row0, tid, red);
    if (tid < 16) outacc[row0 + tid] = s + bo[row0 + tid];
  }
}

__global__ void logsoftmax_1024(const float* __restrict__ in, float* __restrict__ out) {
  __shared__ float sm[256];
  const int tid = threadIdx.x;

  float m = -3.402823466e38f;
  for (int i = tid; i < RNN_OUTPUT; i += 256) m = fmaxf(m, in[i]);
  sm[tid] = m; __syncthreads();
  for (int s = 128; s > 0; s >>= 1) {
    if (tid < s) sm[tid] = fmaxf(sm[tid], sm[tid + s]);
    __syncthreads();
  }
  const float mx = sm[0]; __syncthreads();

  float sum = 0.0f;
  for (int i = tid; i < RNN_OUTPUT; i += 256) sum += __expf(in[i] - mx);
  sm[tid] = sum; __syncthreads();
  for (int s = 128; s > 0; s >>= 1) {
    if (tid < s) sm[tid] += sm[tid + s];
    __syncthreads();
  }
  const float lse = __logf(sm[0]);

  for (int i = tid; i < RNN_OUTPUT; i += 256) out[i] = in[i] - mx - lse;
}

// ---------------------------------------------------------------------------
// Host launcher
// ---------------------------------------------------------------------------
extern "C" void kernel_launch(void* const* d_in, const int* in_sizes, int n_in,
                              void* d_out, int out_size, void* d_ws, size_t ws_size,
                              hipStream_t stream) {
  (void)in_sizes; (void)n_in; (void)out_size; (void)ws_size;

  const float* x    = (const float*)d_in[0];   // [1024,1,1024]
  const float* Wi2h = (const float*)d_in[1];   // [2048,3072]
  const float* bi2h = (const float*)d_in[2];   // [2048]
  const float* Wi2o = (const float*)d_in[3];   // [1024,3072]
  const float* bi2o = (const float*)d_in[4];   // [1024]
  float* out = (float*)d_out;                  // [1,1024]

  // ---- workspace layout ----
  char* ws = (char*)d_ws;
  const size_t nWh = (size_t)RNN_HIDDEN * RNN_COMB;   // 6,291,456 elems
  const size_t nWo = (size_t)RNN_OUTPUT * RNN_COMB;   // 3,145,728 elems
  const size_t nC  = (size_t)RNN_SEQ    * RNN_COMB;   // 3,145,728 elems

  __bf16* Wh_bf = (__bf16*)(ws);
  __bf16* Wo_bf = (__bf16*)(ws + nWh * 2);
  __bf16* cbuf  = (__bf16*)(ws + (nWh + nWo) * 2);
  float*  oacc  = (float*) (ws + (nWh + nWo + nC) * 2);
  int*    bar   = (int*)   ((char*)oacc + RNN_OUTPUT * 4 + 256);

  // ---- per-launch bf16 conversion (weights + activations become L2-resident) ----
  {
    int n = (int)nWh;
    cvt_f32_bf16<<<(n + 255) / 256, 256, 0, stream>>>(Wi2h, Wh_bf, n);
    n = (int)nWo;
    cvt_f32_bf16<<<(n + 255) / 256, 256, 0, stream>>>(Wi2o, Wo_bf, n);
  }
  {
    int n = RNN_SEQ * RNN_COMB;
    init_cbuf<<<(n + 255) / 256, 256, 0, stream>>>(x, cbuf, oacc, bar);
  }

  // ---- persistent sequential RNN with grid-wide barriers ----
  rnn_persistent<<<RNN_BLOCKS, RNN_THREADS, 0, stream>>>(
      Wh_bf, Wo_bf, cbuf, bi2h, bi2o, oacc, bar, bar + 1);

  // ---- final log-softmax over the 1024 logits ----
  logsoftmax_1024<<<1, 256, 0, stream>>>(oacc, out);
}
